// ConvFeatureExtractor_9414568313601
// MI455X (gfx1250) — matrix operand order repaired and checked by reading, not compiled, not run on hardware
//
#include <hip/hip_runtime.h>
#include <stdint.h>

#define KLEN 6
#define NF   8192
#define MK   4096   // 4^6 kmers
#define BB   1024   // batch

#define BM 128
#define BN 128
#define BK 64

typedef __bf16 v16bf __attribute__((ext_vector_type(16)));
typedef float  v8f   __attribute__((ext_vector_type(8)));

// async global->LDS builtin takes pointers to GCC-style int4 vectors:
//   param0: int4 addrspace(1)* (global src), param1: int4 addrspace(3)* (LDS dst)
typedef int async_v4i __attribute__((vector_size(16)));
#define GLOBAL_AS __attribute__((address_space(1)))
#define LDS_AS    __attribute__((address_space(3)))

#if __has_builtin(__builtin_amdgcn_global_load_async_to_lds_b128)
#define USE_ASYNC_LDS 1
#else
#define USE_ASYNC_LDS 0
#endif

static __device__ __forceinline__ void wait_asynccnt0() {
#if __has_builtin(__builtin_amdgcn_s_wait_asynccnt)
    __builtin_amdgcn_s_wait_asynccnt(0);
#else
    asm volatile("s_wait_asynccnt 0x0" ::: "memory");
#endif
}

static __device__ __forceinline__ unsigned short f32_to_bf16(float f) {
    union { float f; uint32_t u; } v; v.f = f;
    uint32_t r = v.u + 0x7FFFu + ((v.u >> 16) & 1u);   // round-to-nearest-even
    return (unsigned short)(r >> 16);
}

// ---------------------------------------------------------------------------
// K1: matches + softmax over kmers -> bf16 probs (NF, MK)
// one block (256 threads) per filter
// ---------------------------------------------------------------------------
__global__ __launch_bounds__(256)
void kmer_softmax_kernel(const float* __restrict__ kmer_params,
                         const float* __restrict__ temperature,
                         unsigned short* __restrict__ probs) {
    const int f = blockIdx.x;
    const int t = threadIdx.x;
    __shared__ float sp[24];      // kmer_params[f] : [4][6]
    __shared__ float red[256];
    if (t < 24) sp[t] = kmer_params[f * 24 + t];
    __syncthreads();
    const float inv_t = 1.0f / temperature[0];

    float m[16];
    float lmax = -3.4e38f;
    #pragma unroll
    for (int it = 0; it < 16; ++it) {
        const int i = t + it * 256;
        float s = 0.f;
        #pragma unroll
        for (int j = 0; j < KLEN; ++j) {
            const int d = (i >> (2 * (KLEN - 1 - j))) & 3;   // base-4 digit, MSB first
            s += sp[d * KLEN + j];
        }
        s *= inv_t;
        m[it] = s;
        lmax = fmaxf(lmax, s);
    }
    red[t] = lmax; __syncthreads();
    for (int off = 128; off > 0; off >>= 1) {
        if (t < off) red[t] = fmaxf(red[t], red[t + off]);
        __syncthreads();
    }
    const float bmax = red[0];
    __syncthreads();

    float lsum = 0.f;
    #pragma unroll
    for (int it = 0; it < 16; ++it) { m[it] = __expf(m[it] - bmax); lsum += m[it]; }
    red[t] = lsum; __syncthreads();
    for (int off = 128; off > 0; off >>= 1) {
        if (t < off) red[t] += red[t + off];
        __syncthreads();
    }
    const float inv_sum = 1.0f / red[0];

    unsigned short* out = probs + (size_t)f * MK;
    #pragma unroll
    for (int it = 0; it < 16; ++it)
        out[t + it * 256] = f32_to_bf16(m[it] * inv_sum);
}

// ---------------------------------------------------------------------------
// K2: fp32 -> bf16 elementwise
// ---------------------------------------------------------------------------
__global__ void f32_to_bf16_kernel(const float* __restrict__ in,
                                   unsigned short* __restrict__ out, int n) {
    const int i = blockIdx.x * blockDim.x + threadIdx.x;
    if (i < n) out[i] = f32_to_bf16(in[i]);
}

// ---------------------------------------------------------------------------
// K3: pooled(BB,NF) = freq_bf16(BB,MK) @ probs_bf16(NF,MK)^T  via WMMA bf16
// block = 256 threads = 8 waves (2 x 4), BM x BN tile, double-buffered LDS,
// async global->LDS fill (GLOBAL_LOAD_ASYNC_TO_LDS_B128 / ASYNCcnt) when
// available so the next tile streams in under the WMMAs.
// ---------------------------------------------------------------------------
__global__ __launch_bounds__(256)
void gemm_kernel(const unsigned short* __restrict__ Abf,   // (BB, MK) bf16 bits
                 const unsigned short* __restrict__ Pbf,   // (NF, MK) bf16 bits
                 float* __restrict__ out) {                // (BB, NF) fp32
    __shared__ unsigned short sA[2][BM * BK];   // 2 x 16 KB
    __shared__ unsigned short sB[2][BN * BK];   // 2 x 16 KB

    const int tid  = threadIdx.x;
    const int lane = tid & 31;
    const int wave = tid >> 5;     // 0..7
    const int wm   = wave >> 2;    // 0..1  -> 64 output rows per wave
    const int wn   = wave & 3;     // 0..3  -> 32 output cols per wave

    const int bm = blockIdx.y * BM;
    const int bn = blockIdx.x * BN;

    const int hrow = lane & 15;
    const int half = lane >> 4;    // lane half selects K sub-range per ISA layout

    v8f acc[4][2];
    const v8f vzero = {0.f, 0.f, 0.f, 0.f, 0.f, 0.f, 0.f, 0.f};
    #pragma unroll
    for (int i = 0; i < 4; ++i)
        #pragma unroll
        for (int j = 0; j < 2; ++j) acc[i][j] = vzero;

    union Frag { v16bf v; uint4 q[2]; };

    // ---- tile fill: 1024 uint4 (16KB) each for A and B ----
    auto load_tile = [&](int kt, int bufIdx) {
        #pragma unroll
        for (int i = 0; i < 4; ++i) {
            const int idx = tid + i * 256;      // 0..1023
            const int r   = idx >> 3;           // tile row 0..127
            const int c   = idx & 7;            // uint4 within row (8 per row)
            const unsigned short* gA = Abf + (size_t)(bm + r) * MK + kt + c * 8;
            const unsigned short* gB = Pbf + (size_t)(bn + r) * MK + kt + c * 8;
#if USE_ASYNC_LDS
            __builtin_amdgcn_global_load_async_to_lds_b128(
                (GLOBAL_AS async_v4i*)gA,
                (LDS_AS async_v4i*)&sA[bufIdx][idx * 8],
                0, 0);
            __builtin_amdgcn_global_load_async_to_lds_b128(
                (GLOBAL_AS async_v4i*)gB,
                (LDS_AS async_v4i*)&sB[bufIdx][idx * 8],
                0, 0);
#else
            reinterpret_cast<uint4*>(sA[bufIdx])[idx] = *reinterpret_cast<const uint4*>(gA);
            reinterpret_cast<uint4*>(sB[bufIdx])[idx] = *reinterpret_cast<const uint4*>(gB);
#endif
        }
    };

    // prologue: fill buffer 0
    load_tile(0, 0);
#if USE_ASYNC_LDS
    wait_asynccnt0();
#endif
    __syncthreads();

    int buf = 0;
    for (int kt = 0; kt < MK; kt += BK) {
        // stream next tile into the other buffer while we compute
        if (kt + BK < MK) load_tile(kt + BK, buf ^ 1);

        // ---- compute: 2 K-steps of 32, 4x2 wmma tiles per wave ----
        #pragma unroll
        for (int ks = 0; ks < BK; ks += 32) {
            Frag a[4], b[2];
            #pragma unroll
            for (int tm = 0; tm < 4; ++tm) {
                // A fragment: lanes 0-15 hold K {0..7,16..23}; lanes 16-31 hold {8..15,24..31}
                const int row = wm * 64 + tm * 16 + hrow;
                const unsigned short* p = &sA[buf][row * BK + ks + half * 8];
                a[tm].q[0] = *reinterpret_cast<const uint4*>(p);
                a[tm].q[1] = *reinterpret_cast<const uint4*>(p + 16);
            }
            #pragma unroll
            for (int tn = 0; tn < 2; ++tn) {
                // B fragment: lane = column; lanes 0-15 hold K {0..7,16..23}; lanes 16-31 hold {8..15,24..31}
                const int col = wn * 32 + tn * 16 + hrow;
                const unsigned short* p = &sB[buf][col * BK + ks + half * 8];
                b[tn].q[0] = *reinterpret_cast<const uint4*>(p);
                b[tn].q[1] = *reinterpret_cast<const uint4*>(p + 16);
            }
            #pragma unroll
            for (int tm = 0; tm < 4; ++tm)
                #pragma unroll
                for (int tn = 0; tn < 2; ++tn)
                    acc[tm][tn] = __builtin_amdgcn_wmma_f32_16x16x32_bf16(
                        false, a[tm].v, false, b[tn].v,
                        (short)0, acc[tm][tn], false, false);
        }

#if USE_ASYNC_LDS
        wait_asynccnt0();       // next buffer fully landed in LDS
#endif
        __syncthreads();        // publish next buffer / retire current one
        buf ^= 1;
    }

    // ---- epilogue: C/D layout -> VGPR r holds M = r + (half ? 8 : 0), N = lane&15
    #pragma unroll
    for (int tm = 0; tm < 4; ++tm) {
        #pragma unroll
        for (int tn = 0; tn < 2; ++tn) {
            const int col     = bn + wn * 32 + tn * 16 + hrow;
            const int rowbase = bm + wm * 64 + tm * 16 + half * 8;
            #pragma unroll
            for (int r = 0; r < 8; ++r)
                out[(size_t)(rowbase + r) * NF + col] = acc[tm][tn][r];
        }
    }
}

// ---------------------------------------------------------------------------
// K4: row-normalize pooled in place: profile = pooled / pooled.sum(axis=1)
// ---------------------------------------------------------------------------
__global__ __launch_bounds__(256)
void normalize_kernel(float* __restrict__ out) {
    const int b = blockIdx.x;
    const int t = threadIdx.x;
    __shared__ float red[256];
    float* row = out + (size_t)b * NF;
    float s = 0.f;
    for (int i = t; i < NF; i += 256) s += row[i];
    red[t] = s; __syncthreads();
    for (int off = 128; off > 0; off >>= 1) {
        if (t < off) red[t] += red[t + off];
        __syncthreads();
    }
    const float inv = 1.0f / red[0];
    for (int i = t; i < NF; i += 256) row[i] *= inv;
}

// ---------------------------------------------------------------------------
extern "C" void kernel_launch(void* const* d_in, const int* in_sizes, int n_in,
                              void* d_out, int out_size, void* d_ws, size_t ws_size,
                              hipStream_t stream) {
    const float* freq    = (const float*)d_in[0];   // (BB, MK)
    const float* kparams = (const float*)d_in[1];   // (NF, 4, 6)
    const float* temp    = (const float*)d_in[2];   // (1,)
    // d_in[3] = kmer_idcs (int32) — digits recomputed arithmetically, unused.
    (void)in_sizes; (void)n_in; (void)out_size; (void)ws_size;

    unsigned short* probs = (unsigned short*)d_ws;            // 64 MB bf16 (NF, MK)
    unsigned short* freqb = probs + (size_t)NF * MK;          //  8 MB bf16 (BB, MK)
    float* out = (float*)d_out;                               // (BB, NF)

    kmer_softmax_kernel<<<NF, 256, 0, stream>>>(kparams, temp, probs);
    f32_to_bf16_kernel<<<(BB * MK) / 256, 256, 0, stream>>>(freq, freqb, BB * MK);
    dim3 grid(NF / BN, BB / BM);   // (64, 8)
    gemm_kernel<<<grid, 256, 0, stream>>>(freqb, probs, out);
    normalize_kernel<<<BB, 256, 0, stream>>>(out);
}
